// SandboxMonolithicBaseline_33346126086688
// MI455X (gfx1250) — compile-verified, matrix operand order checked by reading, not run on hardware
//
#include <hip/hip_runtime.h>
#include <hip/hip_bf16.h>

#define NN 50000
#define NE 800000

typedef __bf16 bf16;
typedef __attribute__((ext_vector_type(16))) __bf16 v16bf;
typedef __attribute__((ext_vector_type(8)))  float  v8f;
typedef __attribute__((ext_vector_type(4)))  unsigned int u32x4;

// ---- gfx1250 async global->LDS via inline asm (bypasses builtin signature
// differences between toolchains).  vdst = LDS byte offset VGPR, vaddr =
// 64-bit global address pair, GV mode.  Tracked by ASYNCcnt.
#if defined(__gfx1250__)
#define HAS_ASYNC_LDS 1
__device__ __forceinline__ void async_ld_b128(const void* gsrc, void* ldst) {
  asm volatile("global_load_async_to_lds_b128 %0, %1, off"
               :
               : "v"((unsigned int)(unsigned long long)ldst),
                 "v"((unsigned long long)gsrc)
               : "memory");
}
__device__ __forceinline__ void wait_async() {
#if __has_builtin(__builtin_amdgcn_s_wait_asynccnt)
  __builtin_amdgcn_s_wait_asynccnt(0);
#else
  asm volatile("s_wait_asynccnt 0x0" ::: "memory");
#endif
}
#else
#define HAS_ASYNC_LDS 0
#endif

// Load a 16x32 bf16 fragment (A or pre-transposed B) from LDS, row-major with
// leading dimension `lda` (elements), K-chunk at `kbase`.  Matches CDNA5
// 16-bit A layout: lanes0-15 rows M=0..15 hold K 0..7 / 16..23, lanes16-31
// hold K 8..15 / 24..31.  Two ds_load_b128 per lane.
__device__ __forceinline__ v16bf load_tile(const bf16* base, int lda, int kbase) {
  const int lane = threadIdx.x & 31;
  const bf16* p = base + (lane & 15) * lda + kbase + ((lane >> 4) << 3);
  union { u32x4 q[2]; v16bf v; } u;
  u.q[0] = *(const u32x4*)(p);
  u.q[1] = *(const u32x4*)(p + 16);
  return u.v;
}

__device__ __forceinline__ v8f wmma_bf16(v16bf a, v16bf b, v8f c) {
  return __builtin_amdgcn_wmma_f32_16x16x32_bf16(false, a, false, b, (short)0, c,
                                                 false, false);
}

// ---------------------------------------------------------------------------
// Kernel 1: node MLP  node_h = relu(relu(ctx27 @ Wn1 + bn1) @ Wn2 + bn2)
//           also zero-fills aggregated[N,64]
// ---------------------------------------------------------------------------
__global__ __launch_bounds__(256) void node_mlp_kernel(
    const float* __restrict__ node_feat, const int* __restrict__ nbi,
    const int* __restrict__ etype, const float* __restrict__ eparams,
    const float* __restrict__ emb,
    const float* __restrict__ Wn1, const float* __restrict__ bn1,
    const float* __restrict__ Wn2, const float* __restrict__ bn2,
    bf16* __restrict__ node_h, float* __restrict__ agg) {
  __shared__ __align__(16) bf16 sW1[64][32];    // Wn1^T, K padded 27->32
  __shared__ __align__(16) bf16 sW2[64][64];    // Wn2^T
  __shared__ float sb1[64], sb2[64];
  __shared__ __align__(16) bf16 sCtx[8][16][32];
  __shared__ __align__(16) bf16 sHid[8][16][64];

  const int tid = threadIdx.x;
  for (int i = tid; i < 64 * 32; i += 256) {
    int n = i >> 5, k = i & 31;
    sW1[n][k] = (bf16)((k < 27) ? Wn1[k * 64 + n] : 0.f);
  }
  for (int i = tid; i < 64 * 64; i += 256) {
    int n = i >> 6, k = i & 63;
    sW2[n][k] = (bf16)Wn2[k * 64 + n];
  }
  if (tid < 64) { sb1[tid] = bn1[tid]; sb2[tid] = bn2[tid]; }

  const int wave = tid >> 5, lane = tid & 31;
  const int node0 = (blockIdx.x * 8 + wave) * 16;
  const int c = lane;
  for (int r = 0; r < 16; ++r) {
    int node = node0 + r;
    float v = 0.f;
    if (node < NN && c < 27) {
      int b = nbi[node];
      if (c < 7)       v = node_feat[node * 7 + c];
      else if (c < 23) v = emb[etype[b] * 16 + (c - 7)];
      else             v = eparams[b * 4 + (c - 23)];
    }
    sCtx[wave][r][c] = (bf16)v;
  }
  __syncthreads();

  const int col0 = lane & 15, mh = (lane >> 4) << 3;
  // layer 1 (K=32)
  v16bf a = load_tile(&sCtx[wave][0][0], 32, 0);
#pragma unroll
  for (int nt = 0; nt < 4; ++nt) {
    v16bf b = load_tile(&sW1[nt * 16][0], 32, 0);
    v8f acc = {};
    acc = wmma_bf16(a, b, acc);
    int col = nt * 16 + col0;
#pragma unroll
    for (int r = 0; r < 8; ++r) {
      float h = acc[r] + sb1[col];
      sHid[wave][r + mh][col] = (bf16)(h > 0.f ? h : 0.f);
    }
  }
  __syncthreads();

  // layer 2 (K=64)
  v16bf a2[2];
  a2[0] = load_tile(&sHid[wave][0][0], 64, 0);
  a2[1] = load_tile(&sHid[wave][0][0], 64, 32);
#pragma unroll
  for (int nt = 0; nt < 4; ++nt) {
    v8f acc = {};
#pragma unroll
    for (int kc = 0; kc < 2; ++kc)
      acc = wmma_bf16(a2[kc], load_tile(&sW2[nt * 16][0], 64, kc * 32), acc);
    int col = nt * 16 + col0;
#pragma unroll
    for (int r = 0; r < 8; ++r) {
      int node = node0 + r + mh;
      if (node < NN) {
        float h = acc[r] + sb2[col];
        node_h[node * 64 + col] = (bf16)(h > 0.f ? h : 0.f);
        agg[node * 64 + col] = 0.f;
      }
    }
  }
}

// ---------------------------------------------------------------------------
// Kernel 2: edge MLP + messages + scatter-add
// Context layout (permuted; We1^T staged with same row permutation):
//   [ node_h[src] 0..63 | node_h[dst] 64..127 | ef 128..131 |
//     emb 132..147 | ep 148..151 | pad 152..159 ]
// ---------------------------------------------------------------------------
__global__ __launch_bounds__(256) void edge_mlp_kernel(
    const int* __restrict__ eidx, const float* __restrict__ edge_feat,
    const int* __restrict__ ebi, const int* __restrict__ etype,
    const float* __restrict__ eparams, const float* __restrict__ emb,
    const bf16* __restrict__ node_h,
    const float* __restrict__ We1, const float* __restrict__ be1,
    const float* __restrict__ We2, const float* __restrict__ be2,
    const float* __restrict__ Wm, const float* __restrict__ bm,
    bf16* __restrict__ edge_h, float* __restrict__ agg) {
  __shared__ __align__(16) bf16 sW1[64][160];
  __shared__ __align__(16) bf16 sW2[64][64];
  __shared__ __align__(16) bf16 sWm[64][64];
  __shared__ float sb1[64], sb2[64], sbm[64];
  __shared__ __align__(16) bf16 sCtx[8][16][160];
  __shared__ __align__(16) bf16 sHid[8][16][64];

  const int tid = threadIdx.x;
  for (int i = tid; i < 64 * 160; i += 256) {
    int n = i / 160, k = i % 160;
    // permuted row mapping: k<128 -> orig k+4 ; 128..131 -> orig k-128 ;
    // 132..151 -> orig k (emb/ep keep position) ; else zero pad
    int orig = (k < 128) ? (k + 4) : (k < 132) ? (k - 128) : (k < 152) ? k : -1;
    sW1[n][k] = (bf16)((orig >= 0) ? We1[orig * 64 + n] : 0.f);
  }
  for (int i = tid; i < 64 * 64; i += 256) {
    int n = i >> 6, k = i & 63;
    sW2[n][k] = (bf16)We2[k * 64 + n];
    sWm[n][k] = (bf16)Wm[k * 64 + n];
  }
  if (tid < 64) { sb1[tid] = be1[tid]; sb2[tid] = be2[tid]; sbm[tid] = bm[tid]; }

  const int wave = tid >> 5, lane = tid & 31;
  const int e0 = (blockIdx.x * 8 + wave) * 16;

#if HAS_ASYNC_LDS
  {
    const int chunk8 = (lane & 7) * 8;   // 16B chunk within a 64-elem row
    const int rsub = lane >> 3;          // 4 rows per issue
#pragma unroll
    for (int i = 0; i < 4; ++i) {
      int r = i * 4 + rsub;
      int e = e0 + r;
      if (e < NE) {
        int s = eidx[e], d = eidx[NE + e];
        async_ld_b128(node_h + s * 64 + chunk8, &sCtx[wave][r][chunk8]);
        async_ld_b128(node_h + d * 64 + chunk8, &sCtx[wave][r][64 + chunk8]);
      } else {
#pragma unroll
        for (int j = 0; j < 8; ++j) {
          sCtx[wave][r][chunk8 + j] = (bf16)0.f;
          sCtx[wave][r][64 + chunk8 + j] = (bf16)0.f;
        }
      }
    }
    // f32-source tail columns 128..159, one column per lane
    int c = 128 + lane;
    for (int r = 0; r < 16; ++r) {
      int e = e0 + r;
      float v = 0.f;
      if (e < NE) {
        int b = ebi[e];
        if (c < 132)      v = edge_feat[e * 4 + (c - 128)];
        else if (c < 148) v = emb[etype[b] * 16 + (c - 132)];
        else if (c < 152) v = eparams[b * 4 + (c - 148)];
      }
      sCtx[wave][r][c] = (bf16)v;
    }
    wait_async();
  }
#else
  for (int r = 0; r < 16; ++r) {
    int e = e0 + r;
    int s = 0, d = 0, b = 0, et = 0;
    bool ok = e < NE;
    if (ok) { s = eidx[e]; d = eidx[NE + e]; b = ebi[e]; et = etype[b]; }
    for (int c = lane; c < 160; c += 32) {
      float v = 0.f;
      if (ok) {
        if (c < 64)       v = (float)node_h[s * 64 + c];
        else if (c < 128) v = (float)node_h[d * 64 + (c - 64)];
        else if (c < 132) v = edge_feat[e * 4 + (c - 128)];
        else if (c < 148) v = emb[et * 16 + (c - 132)];
        else if (c < 152) v = eparams[b * 4 + (c - 148)];
      }
      sCtx[wave][r][c] = (bf16)v;
    }
  }
#endif
  __syncthreads();

  const int col0 = lane & 15, mh = (lane >> 4) << 3;
  // layer 1 (K=160)
  v16bf aT[5];
#pragma unroll
  for (int kc = 0; kc < 5; ++kc) aT[kc] = load_tile(&sCtx[wave][0][0], 160, kc * 32);
#pragma unroll
  for (int nt = 0; nt < 4; ++nt) {
    v8f acc = {};
#pragma unroll
    for (int kc = 0; kc < 5; ++kc)
      acc = wmma_bf16(aT[kc], load_tile(&sW1[nt * 16][0], 160, kc * 32), acc);
    int col = nt * 16 + col0;
#pragma unroll
    for (int r = 0; r < 8; ++r) {
      float h = acc[r] + sb1[col];
      sHid[wave][r + mh][col] = (bf16)(h > 0.f ? h : 0.f);
    }
  }
  __syncthreads();

  // layer 2 (K=64) -> edge_h  (stored to global + reuse sCtx cols 0..63)
  v16bf a2[2];
  a2[0] = load_tile(&sHid[wave][0][0], 64, 0);
  a2[1] = load_tile(&sHid[wave][0][0], 64, 32);
#pragma unroll
  for (int nt = 0; nt < 4; ++nt) {
    v8f acc = {};
#pragma unroll
    for (int kc = 0; kc < 2; ++kc)
      acc = wmma_bf16(a2[kc], load_tile(&sW2[nt * 16][0], 64, kc * 32), acc);
    int col = nt * 16 + col0;
#pragma unroll
    for (int r = 0; r < 8; ++r) {
      int m = r + mh, e = e0 + m;
      float h = acc[r] + sb2[col];
      bf16 hb = (bf16)(h > 0.f ? h : 0.f);
      sCtx[wave][m][col] = hb;
      if (e < NE) edge_h[e * 64 + col] = hb;
    }
  }
  __syncthreads();

  // messages = edge_h @ Wm + bm, scatter-add into both endpoints (L2-resident)
  v16bf a3[2];
  a3[0] = load_tile(&sCtx[wave][0][0], 160, 0);
  a3[1] = load_tile(&sCtx[wave][0][0], 160, 32);
#pragma unroll
  for (int nt = 0; nt < 4; ++nt) {
    v8f acc = {};
#pragma unroll
    for (int kc = 0; kc < 2; ++kc)
      acc = wmma_bf16(a3[kc], load_tile(&sWm[nt * 16][0], 64, kc * 32), acc);
    int col = nt * 16 + col0;
#pragma unroll
    for (int r = 0; r < 8; ++r) {
      int e = e0 + r + mh;
      if (e < NE) {
        float msg = acc[r] + sbm[col];
        int s = eidx[e], d = eidx[NE + e];
        unsafeAtomicAdd(&agg[s * 64 + col], msg);
        unsafeAtomicAdd(&agg[d * 64 + col], msg);
      }
    }
  }
}

// ---------------------------------------------------------------------------
// Kernel 3: node update + node delta
// ---------------------------------------------------------------------------
__global__ __launch_bounds__(256) void node_update_kernel(
    const float* __restrict__ node_feat, const bf16* __restrict__ node_h,
    const float* __restrict__ agg,
    const float* __restrict__ Wu, const float* __restrict__ bu,
    const float* __restrict__ Wnd, const float* __restrict__ bnd,
    bf16* __restrict__ node_hu, float* __restrict__ out_nd,
    float* __restrict__ out_nn) {
  __shared__ __align__(16) bf16 sWu[64][128];   // Wu^T
  __shared__ __align__(16) bf16 sWd[16][64];    // Wnd^T, N padded 7->16
  __shared__ float sbu[64], sbd[16];
  __shared__ __align__(16) bf16 sCtx[8][16][128];
  __shared__ __align__(16) bf16 sHid[8][16][64];

  const int tid = threadIdx.x;
  for (int i = tid; i < 64 * 128; i += 256) {
    int n = i >> 7, k = i & 127;
    sWu[n][k] = (bf16)Wu[k * 64 + n];
  }
  for (int i = tid; i < 16 * 64; i += 256) {
    int n = i >> 6, k = i & 63;
    sWd[n][k] = (bf16)((n < 7) ? Wnd[k * 7 + n] : 0.f);
  }
  if (tid < 64) sbu[tid] = bu[tid];
  if (tid < 16) sbd[tid] = (tid < 7) ? bnd[tid] : 0.f;

  const int wave = tid >> 5, lane = tid & 31;
  const int node0 = (blockIdx.x * 8 + wave) * 16;

#if HAS_ASYNC_LDS
  {
    const int chunk8 = (lane & 7) * 8;
    const int rsub = lane >> 3;
#pragma unroll
    for (int i = 0; i < 4; ++i) {
      int r = i * 4 + rsub;
      int node = node0 + r;
      if (node < NN) {
        async_ld_b128(node_h + node * 64 + chunk8, &sCtx[wave][r][chunk8]);
      } else {
#pragma unroll
        for (int j = 0; j < 8; ++j) sCtx[wave][r][chunk8 + j] = (bf16)0.f;
      }
    }
    for (int r = 0; r < 16; ++r) {
      int node = node0 + r;
      for (int cc = lane; cc < 64; cc += 32) {
        float v = (node < NN) ? agg[node * 64 + cc] : 0.f;
        sCtx[wave][r][64 + cc] = (bf16)v;
      }
    }
    wait_async();
  }
#else
  for (int r = 0; r < 16; ++r) {
    int node = node0 + r;
    bool ok = node < NN;
    for (int c = lane; c < 128; c += 32) {
      float v = 0.f;
      if (ok) v = (c < 64) ? (float)node_h[node * 64 + c] : agg[node * 64 + (c - 64)];
      sCtx[wave][r][c] = (bf16)v;
    }
  }
#endif
  __syncthreads();

  const int col0 = lane & 15, mh = (lane >> 4) << 3;
  // layer 1 (K=128): node_h_updated
  v16bf aT[4];
#pragma unroll
  for (int kc = 0; kc < 4; ++kc) aT[kc] = load_tile(&sCtx[wave][0][0], 128, kc * 32);
#pragma unroll
  for (int nt = 0; nt < 4; ++nt) {
    v8f acc = {};
#pragma unroll
    for (int kc = 0; kc < 4; ++kc)
      acc = wmma_bf16(aT[kc], load_tile(&sWu[nt * 16][0], 128, kc * 32), acc);
    int col = nt * 16 + col0;
#pragma unroll
    for (int r = 0; r < 8; ++r) {
      int m = r + mh, node = node0 + m;
      float h = acc[r] + sbu[col];
      bf16 hb = (bf16)(h > 0.f ? h : 0.f);
      sHid[wave][m][col] = hb;
      if (node < NN) node_hu[node * 64 + col] = hb;
    }
  }
  __syncthreads();

  // layer 2 (K=64, Nout=7 padded to 16): node_delta
  v8f acc = {};
#pragma unroll
  for (int kc = 0; kc < 2; ++kc)
    acc = wmma_bf16(load_tile(&sHid[wave][0][0], 64, kc * 32),
                    load_tile(&sWd[0][0], 64, kc * 32), acc);
#pragma unroll
  for (int r = 0; r < 8; ++r) {
    int node = node0 + r + mh;
    if (node < NN && col0 < 7) {
      float delta = acc[r] + sbd[col0];
      out_nd[node * 7 + col0] = delta;
      out_nn[node * 7 + col0] = node_feat[node * 7 + col0] + delta;
    }
  }
}

// ---------------------------------------------------------------------------
// Kernel 4: edge decoder + edge delta  (all-bf16 context -> fully async)
// ---------------------------------------------------------------------------
__global__ __launch_bounds__(256) void edge_delta_kernel(
    const int* __restrict__ eidx, const float* __restrict__ edge_feat,
    const bf16* __restrict__ edge_h, const bf16* __restrict__ node_hu,
    const float* __restrict__ Wd1, const float* __restrict__ bd1,
    const float* __restrict__ Wd2, const float* __restrict__ bd2,
    float* __restrict__ out_ed, float* __restrict__ out_en) {
  __shared__ __align__(16) bf16 sW1[64][192];   // Wd1^T
  __shared__ __align__(16) bf16 sW2[16][64];    // Wd2^T, N padded 4->16
  __shared__ float sb1[64], sb2[16];
  __shared__ __align__(16) bf16 sCtx[8][16][192];
  __shared__ __align__(16) bf16 sHid[8][16][64];

  const int tid = threadIdx.x;
  for (int i = tid; i < 64 * 192; i += 256) {
    int n = i / 192, k = i % 192;
    sW1[n][k] = (bf16)Wd1[k * 64 + n];
  }
  for (int i = tid; i < 16 * 64; i += 256) {
    int n = i >> 6, k = i & 63;
    sW2[n][k] = (bf16)((n < 4) ? Wd2[k * 4 + n] : 0.f);
  }
  if (tid < 64) sb1[tid] = bd1[tid];
  if (tid < 16) sb2[tid] = (tid < 4) ? bd2[tid] : 0.f;

  const int wave = tid >> 5, lane = tid & 31;
  const int e0 = (blockIdx.x * 8 + wave) * 16;

#if HAS_ASYNC_LDS
  {
    const int chunk8 = (lane & 7) * 8;
    const int rsub = lane >> 3;
#pragma unroll
    for (int i = 0; i < 4; ++i) {
      int r = i * 4 + rsub;
      int e = e0 + r;
      if (e < NE) {
        int s = eidx[e], d = eidx[NE + e];
        async_ld_b128(edge_h + e * 64 + chunk8, &sCtx[wave][r][chunk8]);
        async_ld_b128(node_hu + s * 64 + chunk8, &sCtx[wave][r][64 + chunk8]);
        async_ld_b128(node_hu + d * 64 + chunk8, &sCtx[wave][r][128 + chunk8]);
      } else {
#pragma unroll
        for (int j = 0; j < 8; ++j) {
          sCtx[wave][r][chunk8 + j] = (bf16)0.f;
          sCtx[wave][r][64 + chunk8 + j] = (bf16)0.f;
          sCtx[wave][r][128 + chunk8 + j] = (bf16)0.f;
        }
      }
    }
    wait_async();
  }
#else
  for (int r = 0; r < 16; ++r) {
    int e = e0 + r;
    int s = 0, d = 0;
    bool ok = e < NE;
    if (ok) { s = eidx[e]; d = eidx[NE + e]; }
    for (int c = lane; c < 192; c += 32) {
      float v = 0.f;
      if (ok) {
        if (c < 64)       v = (float)edge_h[e * 64 + c];
        else if (c < 128) v = (float)node_hu[s * 64 + (c - 64)];
        else              v = (float)node_hu[d * 64 + (c - 128)];
      }
      sCtx[wave][r][c] = (bf16)v;
    }
  }
#endif
  __syncthreads();

  const int col0 = lane & 15, mh = (lane >> 4) << 3;
  // layer 1 (K=192)
  v16bf aT[6];
#pragma unroll
  for (int kc = 0; kc < 6; ++kc) aT[kc] = load_tile(&sCtx[wave][0][0], 192, kc * 32);
#pragma unroll
  for (int nt = 0; nt < 4; ++nt) {
    v8f acc = {};
#pragma unroll
    for (int kc = 0; kc < 6; ++kc)
      acc = wmma_bf16(aT[kc], load_tile(&sW1[nt * 16][0], 192, kc * 32), acc);
    int col = nt * 16 + col0;
#pragma unroll
    for (int r = 0; r < 8; ++r) {
      float h = acc[r] + sb1[col];
      sHid[wave][r + mh][col] = (bf16)(h > 0.f ? h : 0.f);
    }
  }
  __syncthreads();

  // layer 2 (K=64, Nout=4 padded to 16)
  v8f acc = {};
#pragma unroll
  for (int kc = 0; kc < 2; ++kc)
    acc = wmma_bf16(load_tile(&sHid[wave][0][0], 64, kc * 32),
                    load_tile(&sW2[0][0], 64, kc * 32), acc);
#pragma unroll
  for (int r = 0; r < 8; ++r) {
    int e = e0 + r + mh;
    if (e < NE && col0 < 4) {
      float delta = acc[r] + sb2[col0];
      out_ed[e * 4 + col0] = delta;
      out_en[e * 4 + col0] = edge_feat[e * 4 + col0] + delta;
    }
  }
}

// ---------------------------------------------------------------------------
extern "C" void kernel_launch(void* const* d_in, const int* in_sizes, int n_in,
                              void* d_out, int out_size, void* d_ws, size_t ws_size,
                              hipStream_t stream) {
  const float* node_feat = (const float*)d_in[0];
  const int*   eidx      = (const int*)d_in[1];
  const float* edge_feat = (const float*)d_in[2];
  const int*   etype     = (const int*)d_in[3];
  const float* eparams   = (const float*)d_in[4];
  const int*   nbi       = (const int*)d_in[5];
  const int*   ebi       = (const int*)d_in[6];
  const float* emb       = (const float*)d_in[9];
  const float* Wn1 = (const float*)d_in[10], *bn1 = (const float*)d_in[11];
  const float* Wn2 = (const float*)d_in[12], *bn2 = (const float*)d_in[13];
  const float* We1 = (const float*)d_in[14], *be1 = (const float*)d_in[15];
  const float* We2 = (const float*)d_in[16], *be2 = (const float*)d_in[17];
  const float* Wm  = (const float*)d_in[18], *bm  = (const float*)d_in[19];
  const float* Wu  = (const float*)d_in[20], *bu  = (const float*)d_in[21];
  const float* Wnd = (const float*)d_in[22], *bnd = (const float*)d_in[23];
  const float* Wd1 = (const float*)d_in[24], *bd1 = (const float*)d_in[25];
  const float* Wd2 = (const float*)d_in[26], *bd2 = (const float*)d_in[27];

  char* ws = (char*)d_ws;
  bf16*  node_h  = (bf16*)(ws);                          // N*64*2  = 6.4 MB
  bf16*  node_hu = (bf16*)(ws + 6400000);                // N*64*2  = 6.4 MB
  float* agg     = (float*)(ws + 12800000);              // N*64*4  = 12.8 MB
  bf16*  edge_h  = (bf16*)(ws + 25600000);               // E*64*2  = 102.4 MB

  float* out = (float*)d_out;
  float* out_nd = out;                       // [N,7]
  float* out_ed = out + NN * 7;              // [E,4]
  float* out_nn = out + NN * 7 + NE * 4;     // [N,7]
  float* out_en = out + 2 * NN * 7 + NE * 4; // [E,4]

  const int node_blocks = (NN + 127) / 128;  // 391
  const int edge_blocks = (NE + 127) / 128;  // 6250

  node_mlp_kernel<<<node_blocks, 256, 0, stream>>>(
      node_feat, nbi, etype, eparams, emb, Wn1, bn1, Wn2, bn2, node_h, agg);
  edge_mlp_kernel<<<edge_blocks, 256, 0, stream>>>(
      eidx, edge_feat, ebi, etype, eparams, emb, node_h,
      We1, be1, We2, be2, Wm, bm, edge_h, agg);
  node_update_kernel<<<node_blocks, 256, 0, stream>>>(
      node_feat, node_h, agg, Wu, bu, Wnd, bnd, node_hu, out_nd, out_nn);
  edge_delta_kernel<<<edge_blocks, 256, 0, stream>>>(
      eidx, edge_feat, edge_h, node_hu, Wd1, bd1, Wd2, bd2, out_ed, out_en);
}
